// EdgeNetwork_20813411516513
// MI455X (gfx1250) — matrix operand order repaired and checked by reading, not compiled, not run on hardware
//
#include <hip/hip_runtime.h>
#include <hip/hip_fp16.h>
#include <hip/hip_bf16.h>

// EdgeNetwork fused kernel for gfx1250 (MI455X), wave32 + WMMA f16->f32.
//
// agg[b,i] = (sum_j attn[i,j] * gelu(u[b,i] + v[b,j] + b1)) @ W2 + b2
//   u[r] = nodes[r] @ W1[:64],  v[r] = nodes[r] @ W1[64:]
// (softmax rows sum to 1 so b2 passes through the weighted sum)

typedef __attribute__((ext_vector_type(16))) _Float16 v16h;
typedef __attribute__((ext_vector_type(8)))  float    v8f;

#define TPB          96      // 3 waves of 32
#define ROWS_PER_WG  48      // 16 batches * 3 nodes; divisible by 3
#define W1T_STRIDE   136     // 128 K-rows + 8 pad halves (bank-conflict pad)
#define W2T_STRIDE   72      // 64 + 8 pad
#define G_STRIDE     72      // 64 + 8 pad

union FragU { v16h v; uint4 q[2]; };

__global__ __launch_bounds__(TPB) void edgenet_fused(
    const float* __restrict__ nodes,   // (3B, 64) flattened
    const float* __restrict__ W1,      // (128, 64)
    const float* __restrict__ b1,      // (64)
    const float* __restrict__ W2,      // (64, 64)
    const float* __restrict__ b2,      // (64)
    const float* __restrict__ eattn,   // (3, 3)
    float* __restrict__ out)           // (3B, 64)
{
    __shared__ __align__(16) _Float16 w1t[64 * W1T_STRIDE]; // [col][k=0..127] transposed
    __shared__ __align__(16) _Float16 w2t[64 * W2T_STRIDE]; // [col][k=0..63]  transposed
    __shared__ __align__(16) float    b1s[64];
    __shared__ __align__(16) float    b2s[64];
    __shared__ __align__(16) float    attns[12];
    __shared__ __align__(16) float    uls[ROWS_PER_WG * 64];
    __shared__ __align__(16) float    vls[ROWS_PER_WG * 64];
    __shared__ __align__(16) _Float16 gls[ROWS_PER_WG * G_STRIDE];

    const int t     = threadIdx.x;
    const int lane  = t & 31;
    const int wave  = t >> 5;
    const int mrow  = lane & 15;            // M (or N) index within tile
    const int hi    = lane >> 4;            // 0 = lanes 0-15, 1 = lanes 16-31
    const int khalfA = hi << 3;             // A-frag K base offset: 0 or 8
    const int klaneB = hi << 4;             // unused: B frags use khalfA, same K base as A

    // ---------- prologue: weights -> LDS (f16, transposed, padded) ----------
    for (int idx = t; idx < 128 * 64; idx += TPB) {
        int k = idx >> 6, c = idx & 63;
        w1t[c * W1T_STRIDE + k] = (_Float16)W1[idx];
    }
    for (int idx = t; idx < 64 * 64; idx += TPB) {
        int k = idx >> 6, c = idx & 63;
        w2t[c * W2T_STRIDE + k] = (_Float16)W2[idx];
    }
    if (t < 64) { b1s[t] = b1[t]; b2s[t] = b2[t]; }
    if (t < 3) { // softmax row t of edge_attn
        float a0 = eattn[t * 3 + 0], a1 = eattn[t * 3 + 1], a2 = eattn[t * 3 + 2];
        float m  = fmaxf(a0, fmaxf(a1, a2));
        float e0 = expf(a0 - m), e1 = expf(a1 - m), e2 = expf(a2 - m);
        float inv = 1.0f / (e0 + e1 + e2);
        attns[t * 3 + 0] = e0 * inv;
        attns[t * 3 + 1] = e1 * inv;
        attns[t * 3 + 2] = e2 * inv;
    }
    __syncthreads();

    // ---------- phase 1: A fragments straight from global (f32 -> f16) ----------
    // 16-bit A 16x32 layout: lane<16 holds row=lane, K in {0..7,16..23};
    // lane>=16 holds row=lane-16, K in {8..15,24..31}. (+32 for second K-block)
    const int row0g = blockIdx.x * ROWS_PER_WG + wave * 16;   // global row of tile
    const float* arow = nodes + (row0g + mrow) * 64;

    v16h afrag[2];
#pragma unroll
    for (int kb = 0; kb < 2; ++kb) {
        const float4 c0 = *(const float4*)(arow + kb * 32 + khalfA + 0);
        const float4 c1 = *(const float4*)(arow + kb * 32 + khalfA + 4);
        const float4 c2 = *(const float4*)(arow + kb * 32 + khalfA + 16);
        const float4 c3 = *(const float4*)(arow + kb * 32 + khalfA + 20);
        v16h f;
        f[0]  = (_Float16)c0.x; f[1]  = (_Float16)c0.y; f[2]  = (_Float16)c0.z; f[3]  = (_Float16)c0.w;
        f[4]  = (_Float16)c1.x; f[5]  = (_Float16)c1.y; f[6]  = (_Float16)c1.z; f[7]  = (_Float16)c1.w;
        f[8]  = (_Float16)c2.x; f[9]  = (_Float16)c2.y; f[10] = (_Float16)c2.z; f[11] = (_Float16)c2.w;
        f[12] = (_Float16)c3.x; f[13] = (_Float16)c3.y; f[14] = (_Float16)c3.z; f[15] = (_Float16)c3.w;
        afrag[kb] = f;
    }

    // ---------- phase 2: U = A@W1a, V = A@W1b via WMMA -> LDS (f32) ----------
    // Dense 16-bit B 32x16 layout: col = lane&15, K in {0..7,16..23} + 8*(lane>=16), same order as A.
    const int rbase = wave * 16 + (hi << 3);   // local row of acc[0]
#pragma unroll
    for (int w = 0; w < 2; ++w) {
        const int koff = w * 64;               // W1a rows 0..63, W1b rows 64..127
        float* dst = w ? vls : uls;
#pragma unroll
        for (int nt = 0; nt < 4; ++nt) {
            const int col = nt * 16 + mrow;
            v8f acc = {};
#pragma unroll
            for (int kb = 0; kb < 2; ++kb) {
                const _Float16* bp = &w1t[col * W1T_STRIDE + koff + kb * 32 + khalfA];
                FragU bf;
                bf.q[0] = *(const uint4*)(bp);
                bf.q[1] = *(const uint4*)(bp + 16);
                acc = __builtin_amdgcn_wmma_f32_16x16x32_f16(
                    false, afrag[kb], false, bf.v, (short)0, acc, false, false);
            }
#pragma unroll
            for (int g = 0; g < 8; ++g)        // C/D layout: row = g + 8*hi
                dst[(rbase + g) * 64 + col] = acc[g];
        }
    }
    __syncthreads();

    // ---------- phase 3: g[r] = sum_j attn[i,j] * gelu(u[r] + v[3b+j] + b1) ----------
    const float inv_sqrt2 = 0.70710678118654752440f;
#pragma unroll 4
    for (int it = 0; it < (ROWS_PER_WG * 64) / TPB; ++it) {
        const int idx = t + it * TPB;
        const int row = idx >> 6;
        const int col = idx & 63;
        const int i   = row % 3;
        const int jb  = row - i;               // first row of this batch (3*b_local)
        const float uval = uls[row * 64 + col] + b1s[col];
        float s = 0.0f;
#pragma unroll
        for (int j = 0; j < 3; ++j) {
            const float x = uval + vls[(jb + j) * 64 + col];
            const float gelu = 0.5f * x * (1.0f + erff(x * inv_sqrt2)); // exact-erf GELU
            s += attns[i * 3 + j] * gelu;
        }
        gls[row * G_STRIDE + col] = (_Float16)s;
    }
    __syncthreads();

    // ---------- phase 4: out = G @ W2 + b2 via WMMA ----------
    v16h gfrag[2];
    const int lrow = wave * 16 + mrow;
#pragma unroll
    for (int kb = 0; kb < 2; ++kb) {
        const _Float16* gp = &gls[lrow * G_STRIDE + kb * 32 + khalfA];
        FragU f;
        f.q[0] = *(const uint4*)(gp);          // halves 0..7  : K = base..base+7
        f.q[1] = *(const uint4*)(gp + 16);     // halves 8..15 : K = base+16..base+23
        gfrag[kb] = f.v;
    }
#pragma unroll
    for (int nt = 0; nt < 4; ++nt) {
        const int col = nt * 16 + mrow;
        v8f acc = {};
#pragma unroll
        for (int kb = 0; kb < 2; ++kb) {
            const _Float16* bp = &w2t[col * W2T_STRIDE + kb * 32 + khalfA];
            FragU bf;
            bf.q[0] = *(const uint4*)(bp);
            bf.q[1] = *(const uint4*)(bp + 16);
            acc = __builtin_amdgcn_wmma_f32_16x16x32_f16(
                false, gfrag[kb], false, bf.v, (short)0, acc, false, false);
        }
        const float bb = b2s[col];
        const int orow = row0g + (hi << 3);
#pragma unroll
        for (int g = 0; g < 8; ++g)
            out[(orow + g) * 64 + col] = acc[g] + bb;
    }
}

extern "C" void kernel_launch(void* const* d_in, const int* in_sizes, int n_in,
                              void* d_out, int out_size, void* d_ws, size_t ws_size,
                              hipStream_t stream) {
    const float* nodes = (const float*)d_in[0];
    const float* W1    = (const float*)d_in[1];
    const float* b1    = (const float*)d_in[2];
    const float* W2    = (const float*)d_in[3];
    const float* b2    = (const float*)d_in[4];
    const float* ea    = (const float*)d_in[5];
    float* out = (float*)d_out;

    const int rows = in_sizes[0] / 64;          // 3*B = 786432
    const int grid = rows / ROWS_PER_WG;        // 16384
    edgenet_fused<<<grid, TPB, 0, stream>>>(nodes, W1, b1, W2, b2, ea, out);
}
